// SchNetEnergyCharge_77017353552169
// MI455X (gfx1250) — matrix operand order repaired and checked by reading, not compiled, not run on hardware
//
#include <hip/hip_runtime.h>
#include <hip/hip_bf16.h>

#define N_NODES 50000
#define N_EDGES 800000
#define N_GRAPH 100
#define HID     128
#define NGAUSS  50
#define KPAD    64
#define NLAYER  6
#define CUTOFF_ 10.0f
#define LOG2_   0.6931471805599453f
#define PI_     3.14159265358979323846f

typedef __attribute__((ext_vector_type(16))) _Float16 v16h;
typedef __attribute__((ext_vector_type(8)))  _Float16 v8h;
typedef __attribute__((ext_vector_type(8)))  float    v8f;
typedef __attribute__((ext_vector_type(4)))  float    v4f;

// ---------------------------------------------------------------- helpers

__device__ __forceinline__ float sspf(float x) {
    // shifted softplus: log(1+e^x) - log(2), overflow-safe
    float sp = (x > 20.f) ? x : __logf(1.f + __expf(x));
    return sp - LOG2_;
}

__device__ __forceinline__ v8f vzero8() {
    v8f z = {0.f, 0.f, 0.f, 0.f, 0.f, 0.f, 0.f, 0.f};
    return z;
}

// A-fragment (16x32 f16) from row-major f16, leading dim `ld` halves.
// ISA layout: lanes 0-15: M=lane, K = 0..7 & 16..23; lanes 16-31: M=lane-16, K = 8..15 & 24..31.
__device__ __forceinline__ v16h load_a16(const _Float16* base, int ld) {
    int lane = threadIdx.x & 31;
    int m  = lane & 15;
    int kb = (lane >> 4) << 3;
    const v8h* p0 = (const v8h*)(base + m * ld + kb);
    const v8h* p1 = (const v8h*)(base + m * ld + kb + 16);
    v8h a = *p0, b = *p1;
    return __builtin_shufflevector(a, b, 0,1,2,3,4,5,6,7,8,9,10,11,12,13,14,15);
}

// A-fragment from row-major f32 (converted to f16 in registers)
__device__ __forceinline__ v16h load_a32(const float* base, int ld) {
    int lane = threadIdx.x & 31;
    int m  = lane & 15;
    int kb = (lane >> 4) << 3;
    const v4f* p0 = (const v4f*)(base + m * ld + kb);
    const v4f* p1 = (const v4f*)(base + m * ld + kb + 16);
    v4f q0 = p0[0], q1 = p0[1], q2 = p1[0], q3 = p1[1];
    v16h r;
#pragma unroll
    for (int i = 0; i < 4; ++i) {
        r[i]      = (_Float16)q0[i];
        r[i + 4]  = (_Float16)q1[i];
        r[i + 8]  = (_Float16)q2[i];
        r[i + 12] = (_Float16)q3[i];
    }
    return r;
}

// B-fragment (32x16 f16) from pre-packed fragment buffer (works for global or LDS ptr):
// one contiguous 32B load per lane.
__device__ __forceinline__ v16h load_b(const _Float16* bf, int ks, int ns, int nfr) {
    int lane = threadIdx.x & 31;
    const v16h* p = (const v16h*)bf + ((ks * nfr + ns) * 32 + lane);
    return *p;
}

__device__ __forceinline__ v8f wmma_f(v16h a, v16h b, v8f c) {
    return __builtin_amdgcn_wmma_f32_16x16x32_f16(false, a, false, b, (short)0, c, false, false);
}

// Async DMA of packed weight fragments into LDS (GLOBAL_LOAD_ASYNC_TO_LDS_B128,
// tracked by ASYNCcnt; bypasses VGPRs). Each enabled lane copies one 16B chunk.
__device__ __forceinline__ void stage_async(_Float16* dst, const _Float16* src, int halves) {
    // low 32 bits of the generic (flat) LDS pointer are the LDS byte address
    unsigned ldsBase = (unsigned)(size_t)dst;
    int nchunk = halves >> 3;  // 16B per chunk
    for (int i = threadIdx.x; i < nchunk; i += (int)blockDim.x) {
        unsigned la = ldsBase + (unsigned)i * 16u;
        const void* ga = (const char*)src + (size_t)i * 16;
        asm volatile("global_load_async_to_lds_b128 %0, %1, off"
                     :: "v"(la), "v"(ga) : "memory");
    }
}

__device__ __forceinline__ void async_wait0() {
    asm volatile("s_wait_asynccnt 0x0" ::: "memory");
}

// ---------------------------------------------------------------- kernels

// Pack row-major f32 [K x Ncols] weight into B-fragment f16 layout; zero-pad rows >= srcK.
__global__ void k_bfrag(const float* __restrict__ src, _Float16* __restrict__ dst,
                        int K, int Ncols, int srcK) {
    int tid = blockIdx.x * blockDim.x + threadIdx.x;
    int total = (K >> 5) * (Ncols >> 4) * 512;
    if (tid >= total) return;
    int i    = tid & 15;
    int lane = (tid >> 4) & 31;
    int f    = tid >> 9;
    int nfr  = Ncols >> 4;
    int ns = f % nfr, ks = f / nfr;
    int k = (ks << 5) + ((lane >> 4) << 3) + (i & 7) + ((i >> 3) << 4);
    int n = (ns << 4) + (lane & 15);
    float v = (k < srcK) ? src[k * Ncols + n] : 0.f;
    dst[tid] = (_Float16)v;
}

// Per-edge: distance, cosine cutoff, Gaussian smearing (f16, padded to KPAD)
__global__ void k_prep(const float* __restrict__ pos, const int* __restrict__ eidx,
                       _Float16* __restrict__ eaf, float* __restrict__ Cc) {
    int e = blockIdx.x * blockDim.x + threadIdx.x;
    if (e >= N_EDGES) return;
    int s = eidx[e], d = eidx[N_EDGES + e];
    float dx = pos[s * 3 + 0] - pos[d * 3 + 0];
    float dy = pos[s * 3 + 1] - pos[d * 3 + 1];
    float dz = pos[s * 3 + 2] - pos[d * 3 + 2];
    float dist = sqrtf(dx * dx + dy * dy + dz * dz);
    Cc[e] = 0.5f * (__cosf(dist * (PI_ / CUTOFF_)) + 1.f);
    const float step  = CUTOFF_ / (float)(NGAUSS - 1);
    const float coeff = -0.5f / (step * step);
    _Float16* row = eaf + (size_t)e * KPAD;
#pragma unroll
    for (int g = 0; g < NGAUSS; ++g) {
        float t = dist - (float)g * step;
        row[g] = (_Float16)__expf(coeff * t * t);
    }
#pragma unroll
    for (int g = NGAUSS; g < KPAD; ++g) row[g] = (_Float16)0.f;
}

// h = emb[atoms], also keep f16 copy
__global__ void k_embed(const int* __restrict__ atoms, const float* __restrict__ emb,
                        float* __restrict__ h, _Float16* __restrict__ hf) {
    int i = blockIdx.x * blockDim.x + threadIdx.x;
    if (i >= N_NODES * HID) return;
    int n = i >> 7, j = i & 127;
    float v = emb[atoms[n] * HID + j];
    h[i] = v;
    hf[i] = (_Float16)v;
}

// xl = hf @ conv_w1   ([N,128] x [128,128], f16 in, f32 out). 16-row tile per wave,
// weights async-staged in LDS, grid-stride over tiles.
__global__ __launch_bounds__(256) void
k_ngemm(const _Float16* __restrict__ hf, const _Float16* __restrict__ wf,
        float* __restrict__ xl) {
    __shared__ __align__(32) _Float16 sw[16384];  // 32KB packed fragments
    stage_async(sw, wf, 16384);
    async_wait0();
    __syncthreads();

    int wave = threadIdx.x >> 5, lane = threadIdx.x & 31;
    int col0 = lane & 15, hi = lane >> 4;
    const int nTiles = N_NODES / 16;              // 3125
    const int strideT = gridDim.x * 8;

    for (int tile = blockIdx.x * 8 + wave; tile < nTiles; tile += strideT) {
        const _Float16* arow = hf + (size_t)tile * 16 * HID;
        // prefetch next A panel (16x128 f16 = 4KB; 32 lanes x 128B)
        if (tile + strideT < nTiles)
            __builtin_prefetch((const char*)(hf + (size_t)(tile + strideT) * 16 * HID)
                               + lane * 128, 0, 0);
        v8f acc[8];
#pragma unroll
        for (int nt = 0; nt < 8; ++nt) acc[nt] = vzero8();
#pragma unroll
        for (int ks = 0; ks < 4; ++ks) {
            v16h a = load_a16(arow + ks * 32, HID);
#pragma unroll
            for (int nt = 0; nt < 8; ++nt)
                acc[nt] = wmma_f(a, load_b(sw, ks, nt, 8), acc[nt]);
        }
#pragma unroll
        for (int nt = 0; nt < 8; ++nt) {
            int col = nt * 16 + col0;
#pragma unroll
            for (int v = 0; v < 8; ++v) {
                int node = tile * 16 + v + hi * 8;
                xl[(size_t)node * HID + col] = acc[nt][v];
            }
        }
    }
}

// Fused edge filter MLP (2 GEMMs) + cutoff + gather(xl[src]) + atomic scatter into agg[dst].
// Weights async-staged in LDS; grid-stride over edge tiles; per-lane prefetch of next panel.
__global__ __launch_bounds__(256) void
k_edge(const _Float16* __restrict__ eaf, const float* __restrict__ Cc,
       const _Float16* __restrict__ w1f, const _Float16* __restrict__ w2f,
       const float* __restrict__ b1, const float* __restrict__ b2,
       const int* __restrict__ eidx, const float* __restrict__ xl,
       float* __restrict__ agg) {
    __shared__ __align__(32) _Float16 sw1[8192];          // 16KB  (64x128 frags)
    __shared__ __align__(32) _Float16 sw2[16384];         // 32KB  (128x128 frags)
    __shared__ __align__(16) _Float16 stile[8][16 * HID]; // 32KB intermediate
    stage_async(sw1, w1f, 8192);
    stage_async(sw2, w2f, 16384);
    async_wait0();
    __syncthreads();

    int wave = threadIdx.x >> 5, lane = threadIdx.x & 31;
    int col0 = lane & 15, hi = lane >> 4;
    _Float16* my = stile[wave];
    const int nTiles = N_EDGES / 16;                      // 50000
    const int strideT = gridDim.x * 8;

    for (int tile = blockIdx.x * 8 + wave; tile < nTiles; tile += strideT) {
        const _Float16* arow = eaf + (size_t)tile * 16 * KPAD;
        // prefetch next tile's edge features (16x64 f16 = 2KB; 32 lanes x 64B)
        if (tile + strideT < nTiles)
            __builtin_prefetch((const char*)(eaf + (size_t)(tile + strideT) * 16 * KPAD)
                               + lane * 64, 0, 0);

        // GEMM1: [16 x 64] x [64 x 128]
        v8f acc[8];
#pragma unroll
        for (int nt = 0; nt < 8; ++nt) acc[nt] = vzero8();
#pragma unroll
        for (int ks = 0; ks < 2; ++ks) {
            v16h a = load_a16(arow + ks * 32, KPAD);
#pragma unroll
            for (int nt = 0; nt < 8; ++nt)
                acc[nt] = wmma_f(a, load_b(sw1, ks, nt, 8), acc[nt]);
        }
#pragma unroll
        for (int nt = 0; nt < 8; ++nt) {
            int col = nt * 16 + col0;
            float bias = b1[col];
#pragma unroll
            for (int v = 0; v < 8; ++v) {
                int r = v + hi * 8;
                my[r * HID + col] = (_Float16)sspf(acc[nt][v] + bias);
            }
        }
        // per-wave LDS RAW: DS ops from one wave complete in order; wait, reload in A layout
        asm volatile("s_wait_dscnt 0x0" ::: "memory");

        // GEMM2: [16 x 128] x [128 x 128]
        v8f acc2[8];
#pragma unroll
        for (int nt = 0; nt < 8; ++nt) acc2[nt] = vzero8();
#pragma unroll
        for (int ks = 0; ks < 4; ++ks) {
            v16h a = load_a16(my + ks * 32, HID);
#pragma unroll
            for (int nt = 0; nt < 8; ++nt)
                acc2[nt] = wmma_f(a, load_b(sw2, ks, nt, 8), acc2[nt]);
        }

        // epilogue: W = (acc2 + b2) * C ; msg = xl[src]*W ; agg[dst] += msg
        int eS[8], eD[8];
        float cC[8];
#pragma unroll
        for (int v = 0; v < 8; ++v) {
            int e = tile * 16 + v + hi * 8;
            eS[v] = eidx[e];
            eD[v] = eidx[N_EDGES + e];
            cC[v] = Cc[e];
        }
#pragma unroll
        for (int nt = 0; nt < 8; ++nt) {
            int col = nt * 16 + col0;
            float bias = b2[col];
#pragma unroll
            for (int v = 0; v < 8; ++v) {
                float wv  = (acc2[nt][v] + bias) * cC[v];
                float msg = xl[(size_t)eS[v] * HID + col] * wv;
                __hip_atomic_fetch_add(&agg[(size_t)eD[v] * HID + col], msg,
                                       __ATOMIC_RELAXED, __HIP_MEMORY_SCOPE_AGENT);
            }
        }
    }
}

// v = ssp(agg@conv_w2 + cb2) @ int_w + int_b ; h += v ; hf = f16(h)
__global__ __launch_bounds__(256) void
k_update(const float* __restrict__ agg,
         const _Float16* __restrict__ cw2f, const float* __restrict__ cb2,
         const _Float16* __restrict__ intwf, const float* __restrict__ intb,
         float* __restrict__ h, _Float16* __restrict__ hf) {
    __shared__ __align__(32) _Float16 swa[16384];         // 32KB conv_w2 frags
    __shared__ __align__(32) _Float16 swb[16384];         // 32KB int_w  frags
    __shared__ __align__(16) _Float16 stile[8][16 * HID]; // 32KB intermediate
    stage_async(swa, cw2f, 16384);
    stage_async(swb, intwf, 16384);
    async_wait0();
    __syncthreads();

    int wave = threadIdx.x >> 5, lane = threadIdx.x & 31;
    int col0 = lane & 15, hi = lane >> 4;
    _Float16* my = stile[wave];
    const int nTiles = N_NODES / 16;                      // 3125
    const int strideT = gridDim.x * 8;

    for (int tile = blockIdx.x * 8 + wave; tile < nTiles; tile += strideT) {
        const float* arow = agg + (size_t)tile * 16 * HID;
        // prefetch next agg panel (16x128 f32 = 8KB; 32 lanes x 256B)
        if (tile + strideT < nTiles)
            __builtin_prefetch((const char*)(agg + (size_t)(tile + strideT) * 16 * HID)
                               + lane * 256, 0, 0);

        v8f acc[8];
#pragma unroll
        for (int nt = 0; nt < 8; ++nt) acc[nt] = vzero8();
#pragma unroll
        for (int ks = 0; ks < 4; ++ks) {
            v16h a = load_a32(arow + ks * 32, HID);
#pragma unroll
            for (int nt = 0; nt < 8; ++nt)
                acc[nt] = wmma_f(a, load_b(swa, ks, nt, 8), acc[nt]);
        }
#pragma unroll
        for (int nt = 0; nt < 8; ++nt) {
            int col = nt * 16 + col0;
            float bias = cb2[col];
#pragma unroll
            for (int v = 0; v < 8; ++v) {
                int r = v + hi * 8;
                my[r * HID + col] = (_Float16)sspf(acc[nt][v] + bias);
            }
        }
        asm volatile("s_wait_dscnt 0x0" ::: "memory");

        v8f acc2[8];
#pragma unroll
        for (int nt = 0; nt < 8; ++nt) acc2[nt] = vzero8();
#pragma unroll
        for (int ks = 0; ks < 4; ++ks) {
            v16h a = load_a16(my + ks * 32, HID);
#pragma unroll
            for (int nt = 0; nt < 8; ++nt)
                acc2[nt] = wmma_f(a, load_b(swb, ks, nt, 8), acc2[nt]);
        }
#pragma unroll
        for (int nt = 0; nt < 8; ++nt) {
            int col = nt * 16 + col0;
            float bias = intb[col];
#pragma unroll
            for (int v = 0; v < 8; ++v) {
                int node = tile * 16 + v + hi * 8;
                size_t idx = (size_t)node * HID + col;
                float nh = h[idx] + acc2[nt][v] + bias;
                h[idx] = nh;
                hf[idx] = (_Float16)nh;
            }
        }
    }
}

// readout head: hh = ssp(h@lin1_w + lin1_b); e/q per atom + per-graph atomic segment sums
__global__ void k_readout(const float* __restrict__ h,
                          const float* __restrict__ w1, const float* __restrict__ bb1,
                          const float* __restrict__ ew, const float* __restrict__ ebp,
                          const float* __restrict__ qw, const float* __restrict__ qbp,
                          const int* __restrict__ batch, float* __restrict__ out) {
    int n = blockIdx.x * blockDim.x + threadIdx.x;
    if (n >= N_NODES) return;
    const float* hr = h + (size_t)n * HID;
    float accv[64];
#pragma unroll
    for (int j = 0; j < 64; ++j) accv[j] = bb1[j];
    for (int k = 0; k < HID; ++k) {
        float hk = hr[k];
        const float* wr = w1 + k * 64;
#pragma unroll
        for (int j = 0; j < 64; ++j) accv[j] += hk * wr[j];
    }
    float e = 0.f, q = 0.f;
#pragma unroll
    for (int j = 0; j < 64; ++j) {
        float s = sspf(accv[j]);
        e += s * ew[j];
        q += s * qw[j];
    }
    e += ebp[0];
    q += qbp[0];
    out[n] = e;
    out[N_NODES + n] = q;
    int b = batch[n];
    __hip_atomic_fetch_add(&out[2 * N_NODES + b], e,
                           __ATOMIC_RELAXED, __HIP_MEMORY_SCOPE_AGENT);
    __hip_atomic_fetch_add(&out[2 * N_NODES + N_GRAPH + b], q,
                           __ATOMIC_RELAXED, __HIP_MEMORY_SCOPE_AGENT);
}

// ---------------------------------------------------------------- host

extern "C" void kernel_launch(void* const* d_in, const int* in_sizes, int n_in,
                              void* d_out, int out_size, void* d_ws, size_t ws_size,
                              hipStream_t stream) {
    (void)in_sizes; (void)n_in; (void)out_size; (void)ws_size;

    const int*   atoms   = (const int*)d_in[0];
    const float* pos     = (const float*)d_in[1];
    const int*   batch   = (const int*)d_in[2];
    const int*   eidx    = (const int*)d_in[3];
    const float* emb     = (const float*)d_in[4];
    const float* mlp_w1  = (const float*)d_in[5];
    const float* mlp_b1  = (const float*)d_in[6];
    const float* mlp_w2  = (const float*)d_in[7];
    const float* mlp_b2  = (const float*)d_in[8];
    const float* conv_w1 = (const float*)d_in[9];
    const float* conv_w2 = (const float*)d_in[10];
    const float* conv_b2 = (const float*)d_in[11];
    const float* int_w   = (const float*)d_in[12];
    const float* int_b   = (const float*)d_in[13];
    const float* lin1_w  = (const float*)d_in[14];
    const float* lin1_b  = (const float*)d_in[15];
    const float* e_w     = (const float*)d_in[16];
    const float* e_b     = (const float*)d_in[17];
    const float* q_w     = (const float*)d_in[18];
    const float* q_b     = (const float*)d_in[19];
    float* out = (float*)d_out;

    // workspace carve (256B aligned regions)
    size_t off = 0;
    char* base = (char*)d_ws;
    auto carve = [&](size_t bytes) -> char* {
        char* p = base + off;
        off += (bytes + 255) & ~(size_t)255;
        return p;
    };
    _Float16* eaf  = (_Float16*)carve((size_t)N_EDGES * KPAD * 2);
    float*    Cc   = (float*)   carve((size_t)N_EDGES * 4);
    float*    h    = (float*)   carve((size_t)N_NODES * HID * 4);
    _Float16* hf   = (_Float16*)carve((size_t)N_NODES * HID * 2);
    float*    xl   = (float*)   carve((size_t)N_NODES * HID * 4);
    float*    agg  = (float*)   carve((size_t)N_NODES * HID * 4);
    _Float16* w1f  = (_Float16*)carve((size_t)NLAYER * 2 * 8 * 512 * 2);  // 64x128 frags
    _Float16* w2f  = (_Float16*)carve((size_t)NLAYER * 4 * 8 * 512 * 2);  // 128x128 frags
    _Float16* cw1f = (_Float16*)carve((size_t)NLAYER * 4 * 8 * 512 * 2);
    _Float16* cw2f = (_Float16*)carve((size_t)NLAYER * 4 * 8 * 512 * 2);
    _Float16* intf = (_Float16*)carve((size_t)NLAYER * 4 * 8 * 512 * 2);

    // pack weights into B-fragment layout (f16)
    for (int l = 0; l < NLAYER; ++l) {
        k_bfrag<<<32, 256, 0, stream>>>(mlp_w1 + (size_t)l * NGAUSS * HID,
                                        w1f + (size_t)l * 8192, 64, 128, NGAUSS);
        k_bfrag<<<64, 256, 0, stream>>>(mlp_w2 + (size_t)l * HID * HID,
                                        w2f + (size_t)l * 16384, 128, 128, 128);
        k_bfrag<<<64, 256, 0, stream>>>(conv_w1 + (size_t)l * HID * HID,
                                        cw1f + (size_t)l * 16384, 128, 128, 128);
        k_bfrag<<<64, 256, 0, stream>>>(conv_w2 + (size_t)l * HID * HID,
                                        cw2f + (size_t)l * 16384, 128, 128, 128);
        k_bfrag<<<64, 256, 0, stream>>>(int_w + (size_t)l * HID * HID,
                                        intf + (size_t)l * 16384, 128, 128, 128);
    }

    k_prep<<<(N_EDGES + 255) / 256, 256, 0, stream>>>(pos, eidx, eaf, Cc);
    k_embed<<<(N_NODES * HID + 255) / 256, 256, 0, stream>>>(atoms, emb, h, hf);
    hipMemsetAsync(out + 2 * N_NODES, 0, (size_t)2 * N_GRAPH * 4, stream);

    const int nodeBlocks = 391;   // 3125 node tiles, 8 waves/block
    const int edgeBlocks = 625;   // 50000 edge tiles, ~10 tiles/wave grid-stride

    for (int l = 0; l < NLAYER; ++l) {
        k_ngemm<<<nodeBlocks, 256, 0, stream>>>(hf, cw1f + (size_t)l * 16384, xl);
        hipMemsetAsync(agg, 0, (size_t)N_NODES * HID * 4, stream);
        k_edge<<<edgeBlocks, 256, 0, stream>>>(eaf, Cc,
                                               w1f + (size_t)l * 8192,
                                               w2f + (size_t)l * 16384,
                                               mlp_b1 + (size_t)l * HID,
                                               mlp_b2 + (size_t)l * HID,
                                               eidx, xl, agg);
        k_update<<<nodeBlocks, 256, 0, stream>>>(agg,
                                                 cw2f + (size_t)l * 16384,
                                                 conv_b2 + (size_t)l * HID,
                                                 intf + (size_t)l * 16384,
                                                 int_b + (size_t)l * HID,
                                                 h, hf);
    }

    k_readout<<<(N_NODES + 255) / 256, 256, 0, stream>>>(h, lin1_w, lin1_b,
                                                         e_w, e_b, q_w, q_b,
                                                         batch, out);
}
